// ToothFormerSeg_8813272891492
// MI455X (gfx1250) — compile-verified, matrix operand order checked by reading, not run on hardware
//
#include <hip/hip_runtime.h>
#include <hip/hip_bf16.h>
#include <math.h>

// ---------------------------------------------------------------------------
// ToothFormerSeg on MI455X (gfx1250).
// Dense linears: wave32 WMMA f16 GEMM, f32 accumulate.
//   - fast path (k_gemm64_async): M%64==0, N%64==0, K%32==0; double-buffered
//     LDS tiles staged with global_load_async_to_lds_{b128,b32} (ASYNCcnt),
//     4 WMMA tiles per wave sharing one A fragment.
//   - guarded path (k_gemm): clamped unconditional loads + select, used only
//     for embed (K=3) and head_fc2 (N=10).
// Irregular kNN / attention / interp: VALU kernels.
// ---------------------------------------------------------------------------

typedef __attribute__((ext_vector_type(16))) _Float16 v16h;
typedef __attribute__((ext_vector_type(8)))  _Float16 v8h;
typedef __attribute__((ext_vector_type(8)))  float    v8f;

#define BC 2
#define P1C 8192
#define P2C 2048
#define P3C 512
#define D1C 128
#define D2C 256
#define D3C 512
#define HEADSC 8

__device__ __forceinline__ float apply_act(float x, int ACT) {
  if (ACT == 1) return fmaxf(x, 0.0f);
  if (ACT == 2) return 0.5f * x * (1.0f + erff(x * 0.70710678118654752f));
  return x;
}

// ---------------------------------------------------------------------------
// Fast WMMA GEMM with async-LDS double buffering.
// Block tile 64x64, 128 threads = 4 waves, each wave computes a 16(M)x64(N)
// strip = 4 WMMA tiles reusing one A fragment. Requires M%64, N%64, K%32 == 0.
// LDS tiles are fp32 (async DMA copies raw bytes); fp32->f16 conversion
// happens during fragment assembly and co-executes with the WMMAs.
// ---------------------------------------------------------------------------
template <int ACT>
__global__ __launch_bounds__(128) void k_gemm64_async(const float* __restrict__ A,
                                                      const float* __restrict__ W,
                                                      const float* __restrict__ bias,
                                                      float* __restrict__ C,
                                                      int M, int N, int K) {
  // row stride 36 floats = 144 B (multiple of 16 -> b128-aligned rows)
  __shared__ float AsF[2][64][36];   // [buf][m][k]
  __shared__ float BsF[2][64][36];   // [buf][n][k]  (transposed W tile)

  const int tid  = threadIdx.x;
  const int lane = tid & 31;
  const int wave = tid >> 5;
  const int bm = blockIdx.y * 64;
  const int bn = blockIdx.x * 64;

  v8f acc[4] = {{}, {}, {}, {}};

  // A staging: thread -> row ar (0..63), col group ac (0/16): 4 x async b128
  const int ar = tid >> 1;
  const int ac = (tid & 1) * 16;
  // W staging (transposed): thread -> k row wr (0..31), cols wc..wc+15:
  // 16 x async b32 scatter (per-lane LDS address = BsF[c][k])
  const int wr = tid >> 2;
  const int wc = (tid & 3) * 16;

  const int half = lane >> 4;
  const int arow = wave * 16 + (lane & 15);

  auto issue_tile = [&](int buf, int kk) {
    const unsigned long long ga =
        (unsigned long long)(A + (size_t)(bm + ar) * K + kk + ac);
    const unsigned la = (unsigned)(size_t)&AsF[buf][ar][ac];
#pragma unroll
    for (int j = 0; j < 4; ++j) {
      asm volatile("global_load_async_to_lds_b128 %0, %1, off"
                   :: "v"(la + 16u * j), "v"(ga + 16ull * j)
                   : "memory");
    }
#pragma unroll
    for (int j = 0; j < 16; ++j) {
      const unsigned long long gw =
          (unsigned long long)(W + (size_t)(kk + wr) * N + bn + wc + j);
      const unsigned lw = (unsigned)(size_t)&BsF[buf][wc + j][wr];
      asm volatile("global_load_async_to_lds_b32 %0, %1, off"
                   :: "v"(lw), "v"(gw)
                   : "memory");
    }
  };

  int buf = 0;
  issue_tile(0, 0);

  for (int kk = 0; kk < K; kk += 32) {
    const bool have_next = (kk + 32 < K);
    if (have_next) issue_tile(buf ^ 1, kk + 32);
    // current tile's 20 async issues complete in order; allow the next
    // tile's 20 to remain outstanding.
    if (have_next) asm volatile("s_wait_asynccnt 0x14" ::: "memory");
    else           asm volatile("s_wait_asynccnt 0x0"  ::: "memory");
    __syncthreads();  // waves consume each other's DMA'd data

    // ---- A fragment: lanes 0-15 hold K 0..7/16..23, lanes 16-31 hold
    //      K 8..15/24..31; fp32 b128 LDS loads + cvt to f16 ----
    const float* arp = &AsF[buf][arow][0];
    const float4 fa0 = *(const float4*)(arp + 8 * half);
    const float4 fa1 = *(const float4*)(arp + 8 * half + 4);
    const float4 fa2 = *(const float4*)(arp + 16 + 8 * half);
    const float4 fa3 = *(const float4*)(arp + 16 + 8 * half + 4);
    const v16h af = {
        (_Float16)fa0.x, (_Float16)fa0.y, (_Float16)fa0.z, (_Float16)fa0.w,
        (_Float16)fa1.x, (_Float16)fa1.y, (_Float16)fa1.z, (_Float16)fa1.w,
        (_Float16)fa2.x, (_Float16)fa2.y, (_Float16)fa2.z, (_Float16)fa2.w,
        (_Float16)fa3.x, (_Float16)fa3.y, (_Float16)fa3.z, (_Float16)fa3.w};

#pragma unroll
    for (int nt = 0; nt < 4; ++nt) {
      const int bcol = nt * 16 + (lane & 15);
      const float* brp = &BsF[buf][bcol][0];
      const float4 fb0 = *(const float4*)(brp + 8 * half);
      const float4 fb1 = *(const float4*)(brp + 8 * half + 4);
      const float4 fb2 = *(const float4*)(brp + 16 + 8 * half);
      const float4 fb3 = *(const float4*)(brp + 16 + 8 * half + 4);
      const v16h bf = {
          (_Float16)fb0.x, (_Float16)fb0.y, (_Float16)fb0.z, (_Float16)fb0.w,
          (_Float16)fb1.x, (_Float16)fb1.y, (_Float16)fb1.z, (_Float16)fb1.w,
          (_Float16)fb2.x, (_Float16)fb2.y, (_Float16)fb2.z, (_Float16)fb2.w,
          (_Float16)fb3.x, (_Float16)fb3.y, (_Float16)fb3.z, (_Float16)fb3.w};
      acc[nt] = __builtin_amdgcn_wmma_f32_16x16x32_f16(false, af, false, bf,
                                                       (short)0, acc[nt],
                                                       false, false);
    }
    __syncthreads();  // all waves done reading buf before it is re-filled
    buf ^= 1;
  }

  // ---- epilogue: C/D layout lanes 0-15 -> M=v, lanes 16-31 -> M=v+8 ----
  const int mb = bm + wave * 16 + (lane >> 4) * 8;
#pragma unroll
  for (int nt = 0; nt < 4; ++nt) {
    const int n = bn + nt * 16 + (lane & 15);
    const float bv = bias ? bias[n] : 0.0f;
#pragma unroll
    for (int v = 0; v < 8; ++v) {
      C[(size_t)(mb + v) * N + n] = apply_act(acc[nt][v] + bv, ACT);
    }
  }
}

// ---------------------------------------------------------------------------
// Guarded WMMA GEMM (edge shapes): 32x32 block tile, clamped unconditional
// loads + value select (avoids exec-mask scalarization).
// ---------------------------------------------------------------------------
template <int ACT>
__global__ __launch_bounds__(128) void k_gemm(const float* __restrict__ A,
                                              const float* __restrict__ W,
                                              const float* __restrict__ bias,
                                              float* __restrict__ C,
                                              int M, int N, int K) {
  __shared__ _Float16 As[32][40];
  __shared__ _Float16 BsT[32][40];  // [n][k]
  const int tid   = threadIdx.x;
  const int lane  = tid & 31;
  const int wave  = tid >> 5;
  const int waveM = wave >> 1;
  const int waveN = wave & 1;
  const int bm = blockIdx.y * 32;
  const int bn = blockIdx.x * 32;

  v8f acc = {};

  const int half = lane >> 4;
  const int arow = waveM * 16 + (lane & 15);
  const int bcol = waveN * 16 + (lane & 15);

  for (int kk = 0; kk < K; kk += 32) {
#pragma unroll
    for (int i = 0; i < 8; ++i) {
      int e = tid + i * 128;  // 0..1023
      int r = e >> 5, c = e & 31;
      int gr = bm + r, gc = kk + c;
      int grc = gr < M ? gr : (M - 1);
      int gcc = gc < K ? gc : (K - 1);
      float av = A[(size_t)grc * K + gcc];
      if (gr >= M || gc >= K) av = 0.0f;
      As[r][c] = (_Float16)av;
      int wrow = kk + r, wcol = bn + c;
      int wrc = wrow < K ? wrow : (K - 1);
      int wcl = wcol < N ? wcol : (N - 1);
      float bv = W[(size_t)wrc * N + wcl];
      if (wrow >= K || wcol >= N) bv = 0.0f;
      BsT[c][r] = (_Float16)bv;
    }
    __syncthreads();

    const v8h alo = *(const v8h*)&As[arow][8 * half];
    const v8h ahi = *(const v8h*)&As[arow][16 + 8 * half];
    const v16h af = __builtin_shufflevector(alo, ahi, 0, 1, 2, 3, 4, 5, 6, 7,
                                            8, 9, 10, 11, 12, 13, 14, 15);
    const v8h blo = *(const v8h*)&BsT[bcol][8 * half];
    const v8h bhi = *(const v8h*)&BsT[bcol][16 + 8 * half];
    const v16h bf = __builtin_shufflevector(blo, bhi, 0, 1, 2, 3, 4, 5, 6, 7,
                                            8, 9, 10, 11, 12, 13, 14, 15);
    acc = __builtin_amdgcn_wmma_f32_16x16x32_f16(false, af, false, bf,
                                                 (short)0, acc, false, false);
    __syncthreads();
  }

  const int n  = bn + waveN * 16 + (lane & 15);
  const int mb = bm + waveM * 16 + (lane >> 4) * 8;
  if (n < N) {
    const float bv = bias ? bias[n] : 0.0f;
#pragma unroll
    for (int v = 0; v < 8; ++v) {
      int m = mb + v;
      if (m < M) C[(size_t)m * N + n] = apply_act(acc[v] + bv, ACT);
    }
  }
}

// ---------------------------------------------------------------------------
// LayerNorm: one wave32 per row (4 rows per 128-thread block)
// ---------------------------------------------------------------------------
__global__ __launch_bounds__(128) void k_layernorm(const float* __restrict__ x,
                                                   const float* __restrict__ g,
                                                   const float* __restrict__ b,
                                                   float* __restrict__ y,
                                                   int rows, int D) {
  const int wave = threadIdx.x >> 5;
  const int lane = threadIdx.x & 31;
  const int row  = blockIdx.x * 4 + wave;
  if (row >= rows) return;
  const float* xr = x + (size_t)row * D;
  float s = 0.0f, s2 = 0.0f;
  for (int d = lane; d < D; d += 32) { float v = xr[d]; s += v; s2 += v * v; }
#pragma unroll
  for (int o = 16; o > 0; o >>= 1) {
    s  += __shfl_xor(s,  o, 32);
    s2 += __shfl_xor(s2, o, 32);
  }
  const float m   = s / (float)D;
  const float var = s2 / (float)D - m * m;
  const float inv = rsqrtf(var + 1e-5f);
  float* yr = y + (size_t)row * D;
  for (int d = lane; d < D; d += 32)
    yr[d] = (xr[d] - m) * inv * g[d] + b[d];
}

// ---------------------------------------------------------------------------
// Brute-force kNN (queries from qxyz against rxyz), K <= 64, insertion sort.
// ---------------------------------------------------------------------------
__global__ __launch_bounds__(64) void k_knn_cross(const float* __restrict__ qxyz,
                                                  const float* __restrict__ rxyz,
                                                  int Bn, int Nq, int Nr, int K,
                                                  int* __restrict__ idx,
                                                  float* __restrict__ dist) {
  const int gid = blockIdx.x * 64 + threadIdx.x;
  if (gid >= Bn * Nq) return;
  const int b = gid / Nq, p = gid % Nq;
  const float* qp = qxyz + ((size_t)b * Nq + p) * 3;
  const float qx = qp[0], qy = qp[1], qz = qp[2];
  const float* rb = rxyz + (size_t)b * Nr * 3;

  float bd[64];
  int   bi[64];
  for (int i = 0; i < K; ++i) { bd[i] = 3.0e38f; bi[i] = 0; }

  for (int j = 0; j < Nr; ++j) {
    const float dx = rb[j * 3 + 0] - qx;
    const float dy = rb[j * 3 + 1] - qy;
    const float dz = rb[j * 3 + 2] - qz;
    const float d2 = dx * dx + dy * dy + dz * dz;
    if (d2 < bd[K - 1]) {
      int pos = K - 1;
      while (pos > 0 && bd[pos - 1] > d2) {
        bd[pos] = bd[pos - 1]; bi[pos] = bi[pos - 1]; --pos;
      }
      bd[pos] = d2; bi[pos] = j;
    }
  }
  for (int i = 0; i < K; ++i) {
    idx[(size_t)gid * K + i] = bi[i];
    if (dist) dist[(size_t)gid * K + i] = sqrtf(fmaxf(bd[i], 0.0f));
  }
}

// ---------------------------------------------------------------------------
// Local attention: one 128-thread workgroup per query point.
// ---------------------------------------------------------------------------
__global__ __launch_bounds__(128) void k_attn(const float* __restrict__ xyz,
                                              const int* __restrict__ idx,
                                              const float* __restrict__ qb,
                                              const float* __restrict__ kb,
                                              const float* __restrict__ vb,
                                              const float* __restrict__ p1w,  // [3,D]
                                              const float* __restrict__ p1b,  // [D]
                                              const float* __restrict__ p2w,  // [D,D]
                                              const float* __restrict__ p2b,  // [D]
                                              float* __restrict__ out,
                                              int Bn, int N, int D, int heads, int K) {
  __shared__ float qs[512];
  __shared__ float hid[512];
  __shared__ float tmp[512];
  __shared__ float lg[512];  // K * heads <= 512

  const int T   = 128;
  const int tid = threadIdx.x;
  const int bp  = blockIdx.x;
  const int b   = bp / N;
  const int p   = bp % N;
  const int dh  = D / heads;
  const float scale = rsqrtf((float)dh);
  const size_t rowq = (size_t)b * N + p;

  const float qx = xyz[rowq * 3 + 0];
  const float qy = xyz[rowq * 3 + 1];
  const float qz = xyz[rowq * 3 + 2];

  for (int d = tid; d < D; d += T) qs[d] = qb[rowq * D + d];
  __syncthreads();

  // ---- pass 1: logits ----
  for (int j = 0; j < K; ++j) {
    const int nj = idx[rowq * K + j];
    if (j + 1 < K) {  // prefetch next neighbor's k row
      const int nn = idx[rowq * K + j + 1];
      __builtin_prefetch(kb + ((size_t)b * N + nn) * D + tid, 0, 1);
    }
    const size_t rown = (size_t)b * N + nj;
    const float rx = xyz[rown * 3 + 0] - qx;
    const float ry = xyz[rown * 3 + 1] - qy;
    const float rz = xyz[rown * 3 + 2] - qz;

    for (int d = tid; d < D; d += T) {
      float h = p1b[d] + rx * p1w[d] + ry * p1w[D + d] + rz * p1w[2 * D + d];
      hid[d] = fmaxf(h, 0.0f);
    }
    __syncthreads();

    const float* krow = kb + rown * D;
    for (int d = tid; d < D; d += T) {
      float pbv = p2b[d];
      for (int i = 0; i < D; ++i) pbv += hid[i] * p2w[(size_t)i * D + d];
      tmp[d] = qs[d] * (krow[d] + pbv);
    }
    __syncthreads();

    if (tid < heads) {
      float s = 0.0f;
      for (int t = 0; t < dh; ++t) s += tmp[tid * dh + t];
      lg[j * heads + tid] = s * scale;
    }
    __syncthreads();
  }

  // ---- per-head softmax over the K neighbors ----
  if (tid < heads) {
    float mx = -3.0e38f;
    for (int j = 0; j < K; ++j) mx = fmaxf(mx, lg[j * heads + tid]);
    float se = 0.0f;
    for (int j = 0; j < K; ++j) {
      float e = expf(lg[j * heads + tid] - mx);
      lg[j * heads + tid] = e;
      se += e;
    }
    const float inv = 1.0f / se;
    for (int j = 0; j < K; ++j) lg[j * heads + tid] *= inv;
  }
  __syncthreads();

  // ---- pass 2: weighted sum of (v + pb) ----
  float accv[4] = {0.0f, 0.0f, 0.0f, 0.0f};
  const int nOwn = D / T;  // 1, 2 or 4
  for (int j = 0; j < K; ++j) {
    const int nj = idx[rowq * K + j];
    if (j + 1 < K) {  // prefetch next neighbor's v row
      const int nn = idx[rowq * K + j + 1];
      __builtin_prefetch(vb + ((size_t)b * N + nn) * D + tid, 0, 1);
    }
    const size_t rown = (size_t)b * N + nj;
    const float rx = xyz[rown * 3 + 0] - qx;
    const float ry = xyz[rown * 3 + 1] - qy;
    const float rz = xyz[rown * 3 + 2] - qz;

    for (int d = tid; d < D; d += T) {
      float h = p1b[d] + rx * p1w[d] + ry * p1w[D + d] + rz * p1w[2 * D + d];
      hid[d] = fmaxf(h, 0.0f);
    }
    __syncthreads();

    const float* vrow = vb + rown * D;
#pragma unroll
    for (int c = 0; c < 4; ++c) {
      if (c < nOwn) {
        const int d = tid + c * T;
        float pbv = p2b[d];
        for (int i = 0; i < D; ++i) pbv += hid[i] * p2w[(size_t)i * D + d];
        const int h = d / dh;
        accv[c] += lg[j * heads + h] * (vrow[d] + pbv);
      }
    }
    __syncthreads();
  }
#pragma unroll
  for (int c = 0; c < 4; ++c)
    if (c < nOwn) out[rowq * D + tid + c * T] = accv[c];
}

// ---------------------------------------------------------------------------
// Downsample gather: linspace(0, Pin-1, Mout) indices; copies feats + xyz.
// ---------------------------------------------------------------------------
__global__ void k_gather_ds(const float* __restrict__ xyz,
                            const float* __restrict__ x,
                            float* __restrict__ xyz_o,
                            float* __restrict__ x_o,
                            int Bn, int Pin, int Mout, int D) {
  const size_t gid = (size_t)blockIdx.x * blockDim.x + threadIdx.x;
  const size_t tot = (size_t)Bn * Mout * D;
  if (gid >= tot) return;
  const int d = (int)(gid % D);
  const size_t bm = gid / D;
  const int m = (int)(bm % Mout);
  const int b = (int)(bm / Mout);
  const int src = (int)((double)m * (double)(Pin - 1) / (double)(Mout - 1));
  x_o[gid] = x[((size_t)b * Pin + src) * D + d];
  if (d < 3)
    xyz_o[((size_t)b * Mout + m) * 3 + d] = xyz[((size_t)b * Pin + src) * 3 + d];
}

// ---------------------------------------------------------------------------
// 3-NN inverse-distance interpolation
// ---------------------------------------------------------------------------
__global__ void k_interp(const float* __restrict__ dist,
                         const int* __restrict__ idx,
                         const float* __restrict__ f2,
                         float* __restrict__ o,
                         int Bn, int N1, int N2, int C) {
  const size_t gid = (size_t)blockIdx.x * blockDim.x + threadIdx.x;
  const size_t tot = (size_t)Bn * N1 * C;
  if (gid >= tot) return;
  const int c = (int)(gid % C);
  const size_t bn = gid / C;
  const int b = (int)(bn / N1);
  const float* dd = dist + bn * 3;
  const int*   ii = idx  + bn * 3;
  const float w0 = 1.0f / fmaxf(dd[0], 1e-8f);
  const float w1 = 1.0f / fmaxf(dd[1], 1e-8f);
  const float w2 = 1.0f / fmaxf(dd[2], 1e-8f);
  const float ws = w0 + w1 + w2;
  const float* fb = f2 + (size_t)b * N2 * C;
  o[gid] = (w0 * fb[(size_t)ii[0] * C + c] +
            w1 * fb[(size_t)ii[1] * C + c] +
            w2 * fb[(size_t)ii[2] * C + c]) / ws;
}

// ---------------------------------------------------------------------------
// Channel concat + residual add
// ---------------------------------------------------------------------------
__global__ void k_concat(const float* __restrict__ a, const float* __restrict__ b,
                         float* __restrict__ o, size_t rows, int Ca, int Cb) {
  const size_t gid = (size_t)blockIdx.x * blockDim.x + threadIdx.x;
  const int C = Ca + Cb;
  const size_t tot = rows * (size_t)C;
  if (gid >= tot) return;
  const int c = (int)(gid % C);
  const size_t r = gid / C;
  o[gid] = (c < Ca) ? a[r * Ca + c] : b[r * Cb + (c - Ca)];
}

__global__ void k_add(float* __restrict__ y, const float* __restrict__ a, size_t n) {
  const size_t gid = (size_t)blockIdx.x * blockDim.x + threadIdx.x;
  if (gid < n) y[gid] += a[gid];
}

// ---------------------------------------------------------------------------
// Host-side GEMM dispatch
// ---------------------------------------------------------------------------
template <int ACT>
static void launch_gemm(const float* A, const float* W, const float* bias,
                        float* C, int M, int N, int K, hipStream_t s) {
  if ((M % 64 == 0) && (N % 64 == 0) && (K % 32 == 0)) {
    k_gemm64_async<ACT><<<dim3(N / 64, M / 64), 128, 0, s>>>(A, W, bias, C,
                                                             M, N, K);
  } else {
    k_gemm<ACT><<<dim3((N + 31) / 32, (M + 31) / 32), 128, 0, s>>>(A, W, bias, C,
                                                                   M, N, K);
  }
}

// ---------------------------------------------------------------------------
// Host orchestration
// ---------------------------------------------------------------------------
extern "C" void kernel_launch(void* const* d_in, const int* in_sizes, int n_in,
                              void* d_out, int out_size, void* d_ws, size_t ws_size,
                              hipStream_t stream) {
  (void)in_sizes; (void)n_in; (void)out_size; (void)ws_size;

  auto F = [&](int i) { return (const float*)d_in[i]; };

  // ---- param leaf indices (jax pytree: dicts flattened in sorted-key order)
  // 0: xyz
  // 1 down1.b  2 down1.w  3 down2.b  4 down2.w  5 embed.b  6 embed.w
  // 7 head_fc1.b 8 head_fc1.w 9 head_fc2.b 10 head_fc2.w 11 head_ln.b 12 head_ln.g
  // 13.. stage1 (2 blocks x 17), 47.. stage2, 81.. stage3
  // 115 up1.b 116 up1.w 117 up2.b 118 up2.w
  // block leaves: +0 fc1.b +1 fc1.w +2 fc2.b +3 fc2.w +4 ln1.b +5 ln1.g
  //               +6 ln2.b +7 ln2.g +8 pos1.b +9 pos1.w +10 pos2.b +11 pos2.w
  //               +12 proj.b +13 proj.w +14 wk +15 wq +16 wv
  const float* xyz = F(0);

  // ---- workspace bump allocator
  char* base = (char*)d_ws;
  size_t off = 0;
  auto AF = [&](size_t nfloats) {
    float* p = (float*)(base + off);
    off = (off + nfloats * 4 + 255) & ~(size_t)255;
    return p;
  };
  auto AI = [&](size_t nints) {
    int* p = (int*)(base + off);
    off = (off + nints * 4 + 255) & ~(size_t)255;
    return p;
  };

  float* x1   = AF((size_t)BC * P1C * D1C);
  float* x2   = AF((size_t)BC * P2C * D2C);
  float* x3   = AF((size_t)BC * P3C * D3C);
  float* xyz2 = AF((size_t)BC * P2C * 3);
  float* xyz3 = AF((size_t)BC * P3C * 3);
  int*   idx1 = AI((size_t)BC * P1C * 32);
  int*   idx2 = AI((size_t)BC * P2C * 48);
  int*   idx3 = AI((size_t)BC * P3C * 64);
  int*   i3   = AI((size_t)BC * P1C * 3);
  float* d3s  = AF((size_t)BC * P1C * 3);
  float* tA   = AF((size_t)BC * P1C * 512);  // widest temp (hidden / concat)
  float* tB   = AF((size_t)BC * P1C * 256);  // interp out / gathered feats
  float* tC   = AF((size_t)BC * P1C * 128);
  float* tD   = AF((size_t)BC * P1C * 128);
  float* tE   = AF((size_t)BC * P1C * 128);
  float* tF   = AF((size_t)BC * P1C * 128);

  auto LN = [&](const float* x, const float* g, const float* b, float* y,
                int rows, int D) {
    k_layernorm<<<(rows + 3) / 4, 128, 0, stream>>>(x, g, b, y, rows, D);
  };
  auto ADD = [&](float* y, const float* a, size_t n) {
    k_add<<<(unsigned)((n + 255) / 256), 256, 0, stream>>>(y, a, n);
  };

  // One transformer block (pre-LN attention + pre-LN MLP, both residual)
  auto BLOCK = [&](int pb, int N, int D, int K, const float* xyzp, const int* idxp,
                   float* x) {
    const int rows = BC * N;
    // attention branch
    LN(x, F(pb + 5), F(pb + 4), tE, rows, D);
    launch_gemm<0>(tE, F(pb + 15), nullptr, tB, rows, D, D, stream);  // q
    launch_gemm<0>(tE, F(pb + 14), nullptr, tC, rows, D, D, stream);  // k
    launch_gemm<0>(tE, F(pb + 16), nullptr, tD, rows, D, D, stream);  // v
    k_attn<<<BC * N, 128, 0, stream>>>(xyzp, idxp, tB, tC, tD,
                                       F(pb + 9), F(pb + 8),   // pos1 w,b
                                       F(pb + 11), F(pb + 10), // pos2 w,b
                                       tF, BC, N, D, HEADSC, K);
    launch_gemm<0>(tF, F(pb + 13), F(pb + 12), tE, rows, D, D, stream);  // proj
    ADD(x, tE, (size_t)rows * D);
    // MLP branch
    LN(x, F(pb + 7), F(pb + 6), tE, rows, D);
    launch_gemm<2>(tE, F(pb + 1), F(pb + 0), tA, rows, D * 4, D, stream);  // fc1+gelu
    launch_gemm<0>(tA, F(pb + 3), F(pb + 2), tE, rows, D, D * 4, stream);  // fc2
    ADD(x, tE, (size_t)rows * D);
  };

  // ---- embed: (B*P1,3) @ (3,128) + b  (guarded path, K=3)
  launch_gemm<0>(xyz, F(6), F(5), x1, BC * P1C, D1C, 3, stream);

  // ---- stage 1
  k_knn_cross<<<(BC * P1C + 63) / 64, 64, 0, stream>>>(xyz, xyz, BC, P1C, P1C, 32,
                                                       idx1, nullptr);
  BLOCK(13, P1C, D1C, 32, xyz, idx1, x1);
  BLOCK(30, P1C, D1C, 32, xyz, idx1, x1);

  // ---- down1: gather linspace rows, then linear 128->256
  {
    size_t tot = (size_t)BC * P2C * D1C;
    k_gather_ds<<<(unsigned)((tot + 255) / 256), 256, 0, stream>>>(
        xyz, x1, xyz2, tB, BC, P1C, P2C, D1C);
    launch_gemm<0>(tB, F(2), F(1), x2, BC * P2C, D2C, D1C, stream);
  }

  // ---- stage 2
  k_knn_cross<<<(BC * P2C + 63) / 64, 64, 0, stream>>>(xyz2, xyz2, BC, P2C, P2C, 48,
                                                       idx2, nullptr);
  BLOCK(47, P2C, D2C, 48, xyz2, idx2, x2);
  BLOCK(64, P2C, D2C, 48, xyz2, idx2, x2);

  // ---- down2: gather + linear 256->512
  {
    size_t tot = (size_t)BC * P3C * D2C;
    k_gather_ds<<<(unsigned)((tot + 255) / 256), 256, 0, stream>>>(
        xyz2, x2, xyz3, tB, BC, P2C, P3C, D2C);
    launch_gemm<0>(tB, F(4), F(3), x3, BC * P3C, D3C, D2C, stream);
  }

  // ---- stage 3
  k_knn_cross<<<(BC * P3C + 63) / 64, 64, 0, stream>>>(xyz3, xyz3, BC, P3C, P3C, 64,
                                                       idx3, nullptr);
  BLOCK(81, P3C, D3C, 64, xyz3, idx3, x3);
  BLOCK(98, P3C, D3C, 64, xyz3, idx3, x3);

  // ---- up2: interp(x3 -> xyz2) ++ x2 -> linear 768->256
  k_knn_cross<<<(BC * P2C + 63) / 64, 64, 0, stream>>>(xyz2, xyz3, BC, P2C, P3C, 3,
                                                       i3, d3s);
  {
    size_t tot = (size_t)BC * P2C * D3C;
    k_interp<<<(unsigned)((tot + 255) / 256), 256, 0, stream>>>(d3s, i3, x3, tB,
                                                                BC, P2C, P3C, D3C);
    size_t tot2 = (size_t)BC * P2C * (D3C + D2C);
    k_concat<<<(unsigned)((tot2 + 255) / 256), 256, 0, stream>>>(
        tB, x2, tA, (size_t)BC * P2C, D3C, D2C);
    launch_gemm<0>(tA, F(118), F(117), tC, BC * P2C, D2C, D3C + D2C, stream);
  }

  // ---- up1: interp(up2 -> xyz) ++ x1 -> linear 384->128
  k_knn_cross<<<(BC * P1C + 63) / 64, 64, 0, stream>>>(xyz, xyz2, BC, P1C, P2C, 3,
                                                       i3, d3s);
  {
    size_t tot = (size_t)BC * P1C * D2C;
    k_interp<<<(unsigned)((tot + 255) / 256), 256, 0, stream>>>(d3s, i3, tC, tB,
                                                                BC, P1C, P2C, D2C);
    size_t tot2 = (size_t)BC * P1C * (D2C + D1C);
    k_concat<<<(unsigned)((tot2 + 255) / 256), 256, 0, stream>>>(
        tB, x1, tA, (size_t)BC * P1C, D2C, D1C);
    launch_gemm<0>(tA, F(116), F(115), tD, BC * P1C, D1C, D2C + D1C, stream);
  }

  // ---- head: LN -> fc1 relu -> fc2 -> d_out (B,P,10)
  LN(tD, F(12), F(11), tE, BC * P1C, D1C);
  launch_gemm<1>(tE, F(8), F(7), tF, BC * P1C, D1C, D1C, stream);
  launch_gemm<0>(tF, F(10), F(9), (float*)d_out, BC * P1C, 10, D1C, stream);
}
